// PointNet2_7842610283209
// MI455X (gfx1250) — matrix operand order repaired and checked by reading, not compiled, not run on hardware
//
#include <hip/hip_runtime.h>
#include <hip/hip_bf16.h>

typedef __attribute__((ext_vector_type(16))) _Float16 v16h;
typedef __attribute__((ext_vector_type(8)))  _Float16 v8h;
typedef __attribute__((ext_vector_type(8)))  float    v8f;

// Problem dims (fixed by the reference setup)
constexpr int B   = 2;
constexpr int N1  = 4096;
constexpr int N2  = 32768;
constexpr int C1  = 128;
constexpr int C2  = 64;
constexpr int CIN = C1 + C2;   // 192
constexpr int F1  = 128;
constexpr int F2  = 128;
constexpr int S0  = 3 + C1;    // inputs0 row stride = 131
constexpr int S1  = 3 + C2;    // inputs1 row stride = 67
constexpr float EPSV = 1e-07f;

// ---------------------------------------------------------------------------
// Kernel 0: convert W1 (192x128) and W2 (128x128) to f16 (row-major kept).
// ---------------------------------------------------------------------------
__global__ __launch_bounds__(256) void cvt_weights(const float* __restrict__ W1,
                                                   const float* __restrict__ W2,
                                                   _Float16* __restrict__ W1h,
                                                   _Float16* __restrict__ W2h) {
    int i = blockIdx.x * 256 + threadIdx.x;
    if (i < CIN * F1) W1h[i] = (_Float16)W1[i];
    if (i < F1 * F2)  W2h[i] = (_Float16)W2[i];
}

// ---------------------------------------------------------------------------
// Kernel 1: brute-force 3-NN over xyz1 (staged in LDS), inverse-distance
// interpolation of points1 features, concat with points2 -> X (f16, 65536x192).
// Also emits xyz2 to the output tail.
// One thread = one query point. xyz1 per batch = 48 KB in LDS; every lane
// reads the same candidate each iteration -> LDS broadcast.
// ---------------------------------------------------------------------------
__global__ __launch_bounds__(256) void knn_interp(const float* __restrict__ in0,
                                                  const float* __restrict__ in1,
                                                  _Float16* __restrict__ X,
                                                  float* __restrict__ outXyz) {
    __shared__ float sx[N1];
    __shared__ float sy[N1];
    __shared__ float sz[N1];

    const int blocksPerBatch = N2 / 256;
    const int b = blockIdx.x / blocksPerBatch;
    const int n = (blockIdx.x % blocksPerBatch) * 256 + threadIdx.x;

    const float* base0 = in0 + (size_t)b * N1 * S0;
    for (int j = threadIdx.x; j < N1; j += 256) {
        const float* p = base0 + (size_t)j * S0;
        sx[j] = p[0];
        sy[j] = p[1];
        sz[j] = p[2];
    }
    __syncthreads();

    const float* q = in1 + ((size_t)b * N2 + n) * S1;
    const float qx = q[0], qy = q[1], qz = q[2];

    float d0 = 3.0e38f, d1 = 3.0e38f, d2 = 3.0e38f;
    int   i0 = 0,       i1 = 0,       i2 = 0;

#pragma unroll 4
    for (int j = 0; j < N1; ++j) {
        const float dx = qx - sx[j];
        const float dy = qy - sy[j];
        const float dz = qz - sz[j];
        const float d  = fmaf(dz, dz, fmaf(dy, dy, dx * dx));
        // strict < keeps the lowest index on ties, matching top_k stability
        if (d < d0) { d2 = d1; i2 = i1; d1 = d0; i1 = i0; d0 = d; i0 = j; }
        else if (d < d1) { d2 = d1; i2 = i1; d1 = d; i1 = j; }
        else if (d < d2) { d2 = d; i2 = j; }
    }

    d0 = fmaxf(d0, EPSV); d1 = fmaxf(d1, EPSV); d2 = fmaxf(d2, EPSV);
    float w0 = 1.0f / d0, w1 = 1.0f / d1, w2 = 1.0f / d2;
    const float inv = 1.0f / (w0 + w1 + w2);
    w0 *= inv; w1 *= inv; w2 *= inv;

    const float* p0 = base0 + (size_t)i0 * S0 + 3;
    const float* p1 = base0 + (size_t)i1 * S0 + 3;
    const float* p2 = base0 + (size_t)i2 * S0 + 3;

    _Float16* xr = X + ((size_t)b * N2 + n) * CIN;
#pragma unroll 8
    for (int c = 0; c < C1; ++c) {
        const float v = fmaf(w2, p2[c], fmaf(w1, p1[c], w0 * p0[c]));
        xr[c] = (_Float16)v;
    }
#pragma unroll 8
    for (int c = 0; c < C2; ++c) {
        xr[C1 + c] = (_Float16)q[3 + c];
    }

    float* xo = outXyz + ((size_t)b * N2 + n) * 3;
    xo[0] = qx; xo[1] = qy; xo[2] = qz;
}

// ---------------------------------------------------------------------------
// Kernel 2: fused 2-layer MLP with v_wmma_f32_16x16x32_f16.
// Block = 256 threads = 8 waves; block owns 16 rows of X.
// Wave w computes output columns [16w, 16w+16).
// Layer 1 (K=192, 6 WMMA) -> bias+ReLU -> f16 H tile in LDS -> barrier ->
// Layer 2 (K=128, 4 WMMA) -> bias+ReLU -> f32 out.
// ---------------------------------------------------------------------------
__global__ __launch_bounds__(256) void mlp_wmma(const _Float16* __restrict__ X,
                                                const _Float16* __restrict__ W1h,
                                                const float* __restrict__ b1,
                                                const _Float16* __restrict__ W2h,
                                                const float* __restrict__ b2,
                                                float* __restrict__ out) {
    __shared__ alignas(16) _Float16 Hs[16 * F1];   // 4 KB layer-1 activations

    const int lane    = threadIdx.x & 31;
    const int wave    = threadIdx.x >> 5;
    const int rowBase = blockIdx.x * 16;
    const int n0      = wave * 16;

    // A-fragment addressing (ISA 16-bit A 16x32 layout):
    //   row M = lane&15 ; K elements: kb + kSel + {0..7} and kb + 16 + kSel + {0..7}
    const int mA   = lane & 15;
    const int kSel = (lane >> 4) << 3;   // 0 or 8
    // C/D addressing: N = n0 + (lane&15); rows r + mOff
    const int nC   = n0 + (lane & 15);
    const int mOff = (lane >> 4) << 3;   // 0 or 8

    // ---- Layer 1: acc = X(16x192) @ W1(192x128), this wave's 16-col stripe
    v8f acc = {};
    const _Float16* Xrow = X + (size_t)(rowBase + mA) * CIN;
#pragma unroll
    for (int c = 0; c < CIN / 32; ++c) {
        const int kb = c * 32;
        union { v16h v; v8h h[2]; } a;
        a.h[0] = *(const v8h*)(Xrow + kb + kSel);
        a.h[1] = *(const v8h*)(Xrow + kb + kSel + 16);
        // B-fragment: lane holds column N = nC, K = kb + kSel + {0..7} and kb + 16 + kSel + {0..7}
        v16h bf;
        for (int s = 0; s < 16; ++s)
            bf[s] = W1h[(size_t)(kb + kSel + (s & 7) + ((s >> 3) << 4)) * F1 + nC];
        acc = __builtin_amdgcn_wmma_f32_16x16x32_f16(
            false, a.v, false, bf, (short)0, acc, false, false);
    }

    const float bias1 = b1[nC];
#pragma unroll
    for (int r = 0; r < 8; ++r) {
        float v = acc[r] + bias1;
        v = v > 0.0f ? v : 0.0f;
        Hs[(r + mOff) * F1 + nC] = (_Float16)v;
    }
    __syncthreads();

    // ---- Layer 2: acc2 = H(16x128) @ W2(128x128), same 16-col stripe
    v8f acc2 = {};
    const _Float16* Hrow = Hs + mA * F1;
#pragma unroll
    for (int c = 0; c < F1 / 32; ++c) {
        const int kb = c * 32;
        union { v16h v; v8h h[2]; } a;
        a.h[0] = *(const v8h*)(Hrow + kb + kSel);
        a.h[1] = *(const v8h*)(Hrow + kb + kSel + 16);
        v16h bf;
        for (int s = 0; s < 16; ++s)
            bf[s] = W2h[(size_t)(kb + kSel + (s & 7) + ((s >> 3) << 4)) * F2 + nC];
        acc2 = __builtin_amdgcn_wmma_f32_16x16x32_f16(
            false, a.v, false, bf, (short)0, acc2, false, false);
    }

    const float bias2 = b2[nC];
#pragma unroll
    for (int r = 0; r < 8; ++r) {
        float v = acc2[r] + bias2;
        v = v > 0.0f ? v : 0.0f;
        out[((size_t)(rowBase + mOff + r)) * F2 + nC] = v;
    }
}

// ---------------------------------------------------------------------------
// Launch
// ---------------------------------------------------------------------------
extern "C" void kernel_launch(void* const* d_in, const int* in_sizes, int n_in,
                              void* d_out, int out_size, void* d_ws, size_t ws_size,
                              hipStream_t stream) {
    const float* in0 = (const float*)d_in[0];   // (B, N1, 131)
    const float* in1 = (const float*)d_in[1];   // (B, N2, 67)
    const float* W1  = (const float*)d_in[2];   // (192, 128)
    const float* b1  = (const float*)d_in[3];   // (128,)
    const float* W2  = (const float*)d_in[4];   // (128, 128)
    const float* b2  = (const float*)d_in[5];   // (128,)

    float* out     = (float*)d_out;
    float* outX    = out;                               // B*N2*F2 floats
    float* outXyz  = out + (size_t)B * N2 * F2;         // B*N2*3 floats

    // Workspace layout
    char* ws = (char*)d_ws;
    _Float16* Xf  = (_Float16*)ws;                                  // 65536 x 192 f16
    const size_t xBytes = (size_t)B * N2 * CIN * sizeof(_Float16);  // ~24 MB
    _Float16* W1h = (_Float16*)(ws + xBytes);
    _Float16* W2h = W1h + (size_t)CIN * F1;

    // 0) weight conversion: max(192*128, 128*128)/256 = 96 blocks
    cvt_weights<<<96, 256, 0, stream>>>(W1, W2, W1h, W2h);

    // 1) kNN + interpolation + concat: B * (N2/256) = 256 blocks
    knn_interp<<<B * (N2 / 256), 256, 0, stream>>>(in0, in1, Xf, outXyz);

    // 2) fused WMMA MLP: (B*N2)/16 = 4096 blocks of 8 waves
    mlp_wmma<<<(B * N2) / 16, 256, 0, stream>>>(Xf, W1h, b1, W2h, b2, outX);
}
